// QuantumSuperpositionAttention_1692217115174
// MI455X (gfx1250) — matrix operand order repaired and checked by reading, not compiled, not run on hardware
//
#include <hip/hip_runtime.h>
#include <math.h>

// QuantumSuperpositionAttention forward for MI455X (gfx1250, wave32, WMMA).
//
// Key simplification: p = |alpha * exp(i*phase)|^2 == |alpha|^2, so the phase
// MLP (inputs w1,b1,w2,b2) is dead code in the forward pass and is skipped.
//
// Cayley: W = (I+A)^-1 (I-A) = (I-A)^2 (I-A^2)^-1, A = S - S^H skew-Hermitian.
// H = I - A^2 is Hermitian PD; invert via Newton-Schulz X <- X(2I - HX),
// X0 = alpha*I with alpha = 2/(1+lambda_max) from a power iteration.
//
// GEMM precision: split-bf16 emulated-f32 (a = a_hi + a_lo, products
// hi*hi + hi*lo + lo*hi on v_wmma_f32_16x16x32_bf16, f32 accumulate).
// LDS holds bf16-PAIR words matching the WMMA VGPR layout exactly, so every
// fragment VGPR is one ds_load_b32 (no v_mov_b16 repacking).

#define N_TOK    1024
#define DMODEL   1024
#define NHEAD    16
#define DHEAD    64
#define BATCH    2
#define KSEL     102
#define SCALE_QK 0.125f
#define NS_ITERS 9
#define POW_ITERS 12

typedef __attribute__((ext_vector_type(16))) __bf16 v16bf;
typedef __attribute__((ext_vector_type(8)))  float  v8f;

union VB16 { v16bf v; unsigned u[8]; };

static __device__ __forceinline__ v8f wmma_bf(v16bf a, v16bf b, v8f c) {
  // D = A(16x32) * B(32x16) + C, f32 accumulate
  return __builtin_amdgcn_wmma_f32_16x16x32_bf16(false, a, false, b, (short)0, c, false, false);
}

// Split two f32 into bf16 (hi,lo) pair-words: hw = hi(f1)<<16 | hi(f0),
// lw = lo(f1)<<16 | lo(f0).  f ~= hi + lo, |lo| <~ 2^-8 |f|.
static __device__ __forceinline__ void pack2(float f0, float f1,
                                             unsigned& hw, unsigned& lw) {
  __bf16 h0 = (__bf16)f0;
  __bf16 h1 = (__bf16)f1;
  __bf16 l0 = (__bf16)(f0 - (float)h0);
  __bf16 l1 = (__bf16)(f1 - (float)h1);
  hw = ((unsigned)__builtin_bit_cast(unsigned short, h1) << 16) |
       (unsigned)__builtin_bit_cast(unsigned short, h0);
  lw = ((unsigned)__builtin_bit_cast(unsigned short, l1) << 16) |
       (unsigned)__builtin_bit_cast(unsigned short, l0);
}

// A-fragment pair-word index: fragment elements (2t,2t+1) hold
// k = 2t + (t>=4 ? 8 : 0) + 8*half  -> word kw = t + (t&4) + 4*half.
#define AKW(t, half) ((t) + ((t) & 4) + ((half) << 2))

// LDS pads: A rows 21 words (gcd(21,64)=1; lane-half +4 offset conflict-free),
// B rows 65 words (bank == k, all 32 lanes distinct).
#define APAD 21
#define BPAD 65

// ---------------------------------------------------------------------------
// Complex GEMM: C = A * B (planar complex f32 in/out, split-bf16 WMMA).
// transB: 0 -> B is (K x N) row-major; 1 -> B is (N x K) row-major (i.e. B^T).
// Block: 256 threads = 8 wave32. Output tile 16(M) x 128(N); wave w owns the
// 16x16 complex tile at N-offset 16w. Dims must be multiples of 16/128/32.
// ---------------------------------------------------------------------------
__global__ __launch_bounds__(256) void cgemm_kernel(
    const float* __restrict__ Are, const float* __restrict__ Aim,
    const float* __restrict__ Bre, const float* __restrict__ Bim,
    float* __restrict__ Cre, float* __restrict__ Cim,
    int M, int N, int K, int lda, int ldb, int ldc, int transB)
{
  // Pair-word planes: [re_hi, re_lo, im_hi, im_lo]
  __shared__ unsigned Aw[4][16][APAD];   // 16 rows x 16 k-pair words
  __shared__ unsigned Bw[4][32][BPAD];   // 32 rows (4 n each) x 64 k-pair words

  const int tid  = threadIdx.x;
  const int lane = tid & 31;
  const int wave = tid >> 5;
  const int half = lane >> 4;   // 0: lanes 0-15, 1: lanes 16-31
  const int mloc = lane & 15;

  const int n0 = blockIdx.x * 128;
  const int m0 = blockIdx.y * 16;

  v8f acc_re = {0.f,0.f,0.f,0.f,0.f,0.f,0.f,0.f};
  v8f acc_im = {0.f,0.f,0.f,0.f,0.f,0.f,0.f,0.f};

  for (int kk = 0; kk < K; kk += 32) {
    __syncthreads();
    // Stage A tile 16x32 (both planes), one pair per thread per plane
    {
      int r = tid >> 4, cw = tid & 15, c = cw << 1;
      size_t g = (size_t)(m0 + r) * lda + kk + c;
      unsigned hw, lw;
      pack2(Are[g], Are[g + 1], hw, lw);
      Aw[0][r][cw] = hw; Aw[1][r][cw] = lw;
      pack2(Aim[g], Aim[g + 1], hw, lw);
      Aw[2][r][cw] = hw; Aw[3][r][cw] = lw;
    }
    // Stage B tile 32x128 (both planes): 2048 pair-words per plane
    if (!transB) {
      for (int i = tid; i < 2048; i += 256) {
        int n = i >> 4, kw = i & 15, k = kw << 1;
        size_t g = (size_t)(kk + k) * ldb + n0 + n;
        unsigned hw, lw;
        pack2(Bre[g], Bre[g + ldb], hw, lw);
        Bw[0][n >> 2][((n & 3) << 4) + kw] = hw; Bw[1][n >> 2][((n & 3) << 4) + kw] = lw;
        pack2(Bim[g], Bim[g + ldb], hw, lw);
        Bw[2][n >> 2][((n & 3) << 4) + kw] = hw; Bw[3][n >> 2][((n & 3) << 4) + kw] = lw;
      }
    } else {
      for (int i = tid; i < 2048; i += 256) {
        int n = i >> 4, kw = i & 15, k = kw << 1;   // consecutive k coalesced
        size_t g0 = (size_t)(n0 + n) * ldb + kk + k;
        size_t g1 = g0 + 1;
        unsigned hw, lw;
        pack2(Bre[g0], Bre[g1], hw, lw);
        Bw[0][n >> 2][((n & 3) << 4) + kw] = hw; Bw[1][n >> 2][((n & 3) << 4) + kw] = lw;
        pack2(Bim[g0], Bim[g1], hw, lw);
        Bw[2][n >> 2][((n & 3) << 4) + kw] = hw; Bw[3][n >> 2][((n & 3) << 4) + kw] = lw;
      }
    }
    // Hint next k-slab into cache while we compute (global_prefetch_b8)
    if (kk + 32 < K) {
      const float* pa = Are + (size_t)(m0 + mloc) * lda + kk + 32;
      __builtin_prefetch(pa, 0, 1);
    }
    __syncthreads();

    // Fragments: each VGPR = one ds_load_b32 (layout-matched pair words)
    VB16 arh, arl, aih, ail, anh, anl, brh, brl, bih, bil;
#pragma unroll
    for (int t = 0; t < 8; ++t) {
      int kw = AKW(t, half);
      unsigned rh = Aw[0][mloc][kw];
      unsigned rl = Aw[1][mloc][kw];
      unsigned ih = Aw[2][mloc][kw];
      unsigned il = Aw[3][mloc][kw];
      arh.u[t] = rh; arl.u[t] = rl;
      aih.u[t] = ih; ail.u[t] = il;
      anh.u[t] = ih ^ 0x80008000u;   // -Ai (NEG bits reserved for bf16)
      anl.u[t] = il ^ 0x80008000u;
    }
    const int nb = wave * 16 + mloc;
#pragma unroll
    for (int t = 0; t < 8; ++t) {
      int kw = ((nb & 3) << 4) + AKW(t, half);
      brh.u[t] = Bw[0][nb >> 2][kw];
      brl.u[t] = Bw[1][nb >> 2][kw];
      bih.u[t] = Bw[2][nb >> 2][kw];
      bil.u[t] = Bw[3][nb >> 2][kw];
    }
    // C_re += Ar*Br - Ai*Bi ; C_im += Ar*Bi + Ai*Br  (3-pass split products)
    acc_re = wmma_bf(arh.v, brh.v, acc_re);
    acc_re = wmma_bf(arh.v, brl.v, acc_re);
    acc_re = wmma_bf(arl.v, brh.v, acc_re);
    acc_re = wmma_bf(anh.v, bih.v, acc_re);
    acc_re = wmma_bf(anh.v, bil.v, acc_re);
    acc_re = wmma_bf(anl.v, bih.v, acc_re);

    acc_im = wmma_bf(arh.v, bih.v, acc_im);
    acc_im = wmma_bf(arh.v, bil.v, acc_im);
    acc_im = wmma_bf(arl.v, bih.v, acc_im);
    acc_im = wmma_bf(aih.v, brh.v, acc_im);
    acc_im = wmma_bf(aih.v, brl.v, acc_im);
    acc_im = wmma_bf(ail.v, brh.v, acc_im);
  }

  // C/D layout: VGPR r, lanes 0-15 -> M=r, lanes 16-31 -> M=8+r; N = lane%16
#pragma unroll
  for (int r = 0; r < 8; ++r) {
    int m = m0 + r + half * 8;
    int n = n0 + wave * 16 + mloc;
    Cre[(size_t)m * ldc + n] = acc_re[r];
    Cim[(size_t)m * ldc + n] = acc_im[r];
  }
}

// ---------------------------------------------------------------------------
// Attention scores: p[b,h,n,s] = |SCALE * <conj(Q[b,h,n,:]), K[b,h,s,:]>|^2
// Q/K planar (B*N, DMODEL); head h = columns [h*64, h*64+64).
// Tile: 16 queries x 128 keys per block, K-dim = 64 (two 32-steps).
// ---------------------------------------------------------------------------
__global__ __launch_bounds__(256) void scores_kernel(
    const float* __restrict__ Qre, const float* __restrict__ Qim,
    const float* __restrict__ Kre, const float* __restrict__ Kim,
    float* __restrict__ P)
{
  __shared__ unsigned Qw[4][16][APAD];
  __shared__ unsigned Kw[4][32][BPAD];

  const int tid  = threadIdx.x;
  const int lane = tid & 31;
  const int wave = tid >> 5;
  const int half = lane >> 4;
  const int mloc = lane & 15;

  const int s0 = blockIdx.x * 128;   // key tile
  const int m0 = blockIdx.y * 16;    // query tile
  const int bh = blockIdx.z;         // b*NHEAD + h
  const int b  = bh >> 4;
  const int h  = bh & 15;

  v8f accr = {0.f,0.f,0.f,0.f,0.f,0.f,0.f,0.f};
  v8f acci = {0.f,0.f,0.f,0.f,0.f,0.f,0.f,0.f};

  for (int kk = 0; kk < DHEAD; kk += 32) {
    __syncthreads();
    {
      int r = tid >> 4, cw = tid & 15, c = cw << 1;
      size_t g = ((size_t)b * N_TOK + m0 + r) * DMODEL + h * DHEAD + kk + c;
      unsigned hw, lw;
      pack2(Qre[g], Qre[g + 1], hw, lw);
      Qw[0][r][cw] = hw; Qw[1][r][cw] = lw;
      pack2(Qim[g], Qim[g + 1], hw, lw);
      Qw[2][r][cw] = hw; Qw[3][r][cw] = lw;
    }
    for (int i = tid; i < 2048; i += 256) {
      int n = i >> 4, kw = i & 15, k = kw << 1;   // transposed stage
      size_t g0 = ((size_t)b * N_TOK + s0 + n) * DMODEL + h * DHEAD + kk + k;
      size_t g1 = g0 + 1;
      unsigned hw, lw;
      pack2(Kre[g0], Kre[g1], hw, lw);
      Kw[0][n >> 2][((n & 3) << 4) + kw] = hw; Kw[1][n >> 2][((n & 3) << 4) + kw] = lw;
      pack2(Kim[g0], Kim[g1], hw, lw);
      Kw[2][n >> 2][((n & 3) << 4) + kw] = hw; Kw[3][n >> 2][((n & 3) << 4) + kw] = lw;
    }
    __syncthreads();

    VB16 qrh, qrl, qih, qil, qnh, qnl, krh, krl, kih, kil;
#pragma unroll
    for (int t = 0; t < 8; ++t) {
      int kw = AKW(t, half);
      unsigned rh = Qw[0][mloc][kw];
      unsigned rl = Qw[1][mloc][kw];
      unsigned ih = Qw[2][mloc][kw];
      unsigned il = Qw[3][mloc][kw];
      qrh.u[t] = rh; qrl.u[t] = rl;
      qih.u[t] = ih; qil.u[t] = il;
      qnh.u[t] = ih ^ 0x80008000u;
      qnl.u[t] = il ^ 0x80008000u;
    }
    const int nb = wave * 16 + mloc;
#pragma unroll
    for (int t = 0; t < 8; ++t) {
      int kw = ((nb & 3) << 4) + AKW(t, half);
      krh.u[t] = Kw[0][nb >> 2][kw];
      krl.u[t] = Kw[1][nb >> 2][kw];
      kih.u[t] = Kw[2][nb >> 2][kw];
      kil.u[t] = Kw[3][nb >> 2][kw];
    }
    // conj(Q).K : re = qr*kr + qi*ki ; im = qr*ki - qi*kr
    accr = wmma_bf(qrh.v, krh.v, accr);
    accr = wmma_bf(qrh.v, krl.v, accr);
    accr = wmma_bf(qrl.v, krh.v, accr);
    accr = wmma_bf(qih.v, kih.v, accr);
    accr = wmma_bf(qih.v, kil.v, accr);
    accr = wmma_bf(qil.v, kih.v, accr);

    acci = wmma_bf(qrh.v, kih.v, acci);
    acci = wmma_bf(qrh.v, kil.v, acci);
    acci = wmma_bf(qrl.v, kih.v, acci);
    acci = wmma_bf(qnh.v, krh.v, acci);
    acci = wmma_bf(qnh.v, krl.v, acci);
    acci = wmma_bf(qnl.v, krh.v, acci);
  }

#pragma unroll
  for (int r = 0; r < 8; ++r) {
    int qm = m0 + r + half * 8;
    int sn = s0 + wave * 16 + mloc;
    float vr = accr[r] * SCALE_QK;
    float vi = acci[r] * SCALE_QK;
    P[((size_t)bh * N_TOK + qm) * N_TOK + sn] = vr * vr + vi * vi;
  }
}

// ---------------------------------------------------------------------------
// Per-row: attn = p/(sum p + 1e-9); top-102 via bitonic sort in LDS; renorm;
// softmax(topk*102); out[d] = sum_j w_j * V[idx_j, d]. One block per row.
// ---------------------------------------------------------------------------
__global__ __launch_bounds__(256) void topk_gather_kernel(
    const float* __restrict__ P,
    const float* __restrict__ Vre, const float* __restrict__ Vim,
    float* __restrict__ Ore, float* __restrict__ Oim)
{
  __shared__ float vals[1024];
  __shared__ int   idxs[1024];
  __shared__ float red[256];
  __shared__ float wsh[128];

  const int tid = threadIdx.x;
  const int row = blockIdx.x;            // (b*NHEAD + h)*N_TOK + n
  const int b   = row >> 14;
  const int h   = (row >> 10) & 15;
  const int n   = row & 1023;
  const float* prow = P + (size_t)row * N_TOK;

  // Load + row sum
  float ls = 0.f;
  for (int i = tid; i < 1024; i += 256) {
    float v = prow[i];
    vals[i] = v;
    idxs[i] = i;
    ls += v;
  }
  red[tid] = ls;
  __syncthreads();
  for (int s = 128; s > 0; s >>= 1) {
    if (tid < s) red[tid] += red[tid + s];
    __syncthreads();
  }
  float inv = 1.0f / (red[0] + 1e-9f);
  __syncthreads();
  for (int i = tid; i < 1024; i += 256) vals[i] *= inv;   // attn
  __syncthreads();

  // Bitonic sort, descending (value, index) pairs
  for (int ksz = 2; ksz <= 1024; ksz <<= 1) {
    for (int j = ksz >> 1; j > 0; j >>= 1) {
      for (int t = tid; t < 1024; t += 256) {
        int ixj = t ^ j;
        if (ixj > t) {
          bool asc = ((t & ksz) != 0);     // descending overall
          float a = vals[t], c = vals[ixj];
          bool sw = asc ? (a > c) : (a < c);
          if (sw) {
            vals[t] = c; vals[ixj] = a;
            int ti = idxs[t]; idxs[t] = idxs[ixj]; idxs[ixj] = ti;
          }
        }
      }
      __syncthreads();
    }
  }

  // Sum of top-102
  red[tid] = (tid < KSEL) ? vals[tid] : 0.f;
  __syncthreads();
  for (int s = 128; s > 0; s >>= 1) {
    if (tid < s) red[tid] += red[tid + s];
    __syncthreads();
  }
  float denom = red[0] + 1e-8f;
  __syncthreads();

  // softmax over t_j = KSEL * vals[j]/denom  (vals sorted desc, max = j0)
  float tmax = (float)KSEL * vals[0] / denom;
  float e = 0.f;
  if (tid < KSEL) e = __expf((float)KSEL * vals[tid] / denom - tmax);
  if (tid < 128) wsh[tid] = e;
  red[tid] = (tid < KSEL) ? e : 0.f;
  __syncthreads();
  for (int s = 128; s > 0; s >>= 1) {
    if (tid < s) red[tid] += red[tid + s];
    __syncthreads();
  }
  float winv = 1.0f / red[0];
  __syncthreads();
  if (tid < KSEL) wsh[tid] *= winv;
  __syncthreads();

  // Gather-accumulate V: threads 0..63 real plane, 64..127 imag plane
  if (tid < 128) {
    int d = tid & 63;
    const float* Vp = (tid < 64) ? Vre : Vim;
    float*       Op = (tid < 64) ? Ore : Oim;
    size_t vbase = (size_t)b * N_TOK * DMODEL + h * DHEAD + d;
    float acc = 0.f;
    for (int j = 0; j < KSEL; ++j)
      acc += wsh[j] * Vp[vbase + (size_t)idxs[j] * DMODEL];
    Op[((size_t)b * N_TOK + n) * DMODEL + h * DHEAD + d] = acc;
  }
}

// ------------------------- small elementwise kernels -----------------------
__global__ __launch_bounds__(256) void buildA_kernel(
    const float* __restrict__ Sre, const float* __restrict__ Sim,
    float* __restrict__ Are, float* __restrict__ Aim)
{
  size_t idx = (size_t)blockIdx.x * 256 + threadIdx.x;
  int i = (int)(idx >> 10), j = (int)(idx & 1023);
  size_t tr = (size_t)j * 1024 + i;
  Are[idx] = Sre[idx] - Sre[tr];   // A = S - S^H
  Aim[idx] = Sim[idx] + Sim[tr];
}

__global__ __launch_bounds__(256) void prepare_kernel(
    const float* __restrict__ Are, const float* __restrict__ Aim,
    const float* __restrict__ A2re, const float* __restrict__ A2im,
    float* __restrict__ Hre, float* __restrict__ Him,
    float* __restrict__ B2re, float* __restrict__ B2im)
{
  size_t idx = (size_t)blockIdx.x * 256 + threadIdx.x;
  int i = (int)(idx >> 10), j = (int)(idx & 1023);
  float d = (i == j) ? 1.f : 0.f;
  Hre[idx]  = d - A2re[idx];                       // H = I - A^2
  Him[idx]  = -A2im[idx];
  B2re[idx] = d - 2.f * Are[idx] + A2re[idx];      // (I-A)^2
  B2im[idx] = -2.f * Aim[idx] + A2im[idx];
}

__global__ __launch_bounds__(256) void uker_kernel(
    const float* __restrict__ Tre, const float* __restrict__ Tim,
    float* __restrict__ Ure, float* __restrict__ Uim)
{
  size_t idx = (size_t)blockIdx.x * 256 + threadIdx.x;
  int i = (int)(idx >> 10), j = (int)(idx & 1023);
  Ure[idx] = ((i == j) ? 2.f : 0.f) - Tre[idx];    // U = 2I - T
  Uim[idx] = -Tim[idx];
}

__global__ __launch_bounds__(256) void initX_kernel(
    float* __restrict__ Xre, float* __restrict__ Xim, const float* __restrict__ scal)
{
  size_t idx = (size_t)blockIdx.x * 256 + threadIdx.x;
  int i = (int)(idx >> 10), j = (int)(idx & 1023);
  Xre[idx] = (i == j) ? scal[1] : 0.f;
  Xim[idx] = 0.f;
}

__global__ __launch_bounds__(256) void initx_kernel(float* xr, float* xi) {
  int i = blockIdx.x * 256 + threadIdx.x;
  if (i < 1024) { xr[i] = 0.03125f; xi[i] = 0.f; }   // unit norm
}

__global__ __launch_bounds__(256) void cmatvec_kernel(
    const float* __restrict__ Hre, const float* __restrict__ Him,
    const float* __restrict__ xr, const float* __restrict__ xi,
    float* __restrict__ yr, float* __restrict__ yi)
{
  int r = blockIdx.x * 256 + threadIdx.x;
  if (r >= 1024) return;
  float ar = 0.f, ai = 0.f;
  size_t base = (size_t)r * 1024;
  for (int c = 0; c < 1024; ++c) {
    float hr = Hre[base + c], hi = Him[base + c];
    float vr = xr[c], vi = xi[c];
    ar += hr * vr - hi * vi;
    ai += hr * vi + hi * vr;
  }
  yr[r] = ar; yi[r] = ai;
}

__global__ __launch_bounds__(256) void normscale_kernel(
    const float* __restrict__ yr, const float* __restrict__ yi,
    float* __restrict__ xr, float* __restrict__ xi, float* __restrict__ scal)
{
  __shared__ float red[256];
  int tid = threadIdx.x;
  float s = 0.f;
  for (int i = tid; i < 1024; i += 256) s += yr[i]*yr[i] + yi[i]*yi[i];
  red[tid] = s;
  __syncthreads();
  for (int st = 128; st > 0; st >>= 1) {
    if (tid < st) red[tid] += red[tid + st];
    __syncthreads();
  }
  float lam = sqrtf(red[0] + 1e-30f);
  float inv = 1.0f / lam;
  for (int i = tid; i < 1024; i += 256) { xr[i] = yr[i]*inv; xi[i] = yi[i]*inv; }
  if (tid == 0) scal[0] = lam;
}

__global__ void alpha_kernel(float* scal) {
  scal[1] = 2.0f / (1.0f + 1.05f * scal[0] + 1e-6f);   // alpha = 2/(1+hmax)
}

__global__ __launch_bounds__(256) void interleave_kernel(
    const float* __restrict__ re, const float* __restrict__ im, float* __restrict__ out)
{
  size_t i = (size_t)blockIdx.x * 256 + threadIdx.x;
  out[2*i]   = re[i];
  out[2*i+1] = im[i];
}

// ---------------------------------------------------------------------------
static void launch_cgemm(const float* Are, const float* Aim,
                         const float* Bre, const float* Bim,
                         float* Cre, float* Cim,
                         int M, int N, int K, int lda, int ldb, int ldc,
                         int transB, hipStream_t stream)
{
  dim3 grid(N / 128, M / 16);
  cgemm_kernel<<<grid, 256, 0, stream>>>(Are, Aim, Bre, Bim, Cre, Cim,
                                         M, N, K, lda, ldb, ldc, transB);
}

extern "C" void kernel_launch(void* const* d_in, const int* in_sizes, int n_in,
                              void* d_out, int out_size, void* d_ws, size_t ws_size,
                              hipStream_t stream)
{
  (void)in_sizes; (void)n_in; (void)out_size; (void)ws_size;

  const float* x_re = (const float*)d_in[0];
  const float* x_im = (const float*)d_in[1];
  // d_in[2..9] = Sq_re,Sq_im,Sk_re,Sk_im,Sv_re,Sv_im,So_re,So_im
  // d_in[10..13] = w1,b1,w2,b2 : dead code in forward (|e^{i phi}| == 1)

  const size_t MM = (size_t)1024 * 1024;
  float* base = (float*)d_ws;
  size_t off = 0;
  auto alloc = [&](size_t nf) { float* p = base + off; off += nf; return p; };

  float* Wm[4][2];
  for (int i = 0; i < 4; ++i) { Wm[i][0] = alloc(MM); Wm[i][1] = alloc(MM); }
  float* Are  = alloc(MM); float* Aim  = alloc(MM);
  float* Tre  = alloc(MM); float* Tim  = alloc(MM);   // also holds A^2
  float* Hre  = alloc(MM); float* Him  = alloc(MM);
  float* B2re = alloc(MM); float* B2im = alloc(MM);
  float* Xre  = alloc(MM); float* Xim  = alloc(MM);
  float* X2re = alloc(MM); float* X2im = alloc(MM);
  float* Ure  = alloc(MM); float* Uim  = alloc(MM);
  float* Qre  = alloc(2*MM); float* Qim = alloc(2*MM);
  float* Kre  = alloc(2*MM); float* Kim = alloc(2*MM);
  float* Vre  = alloc(2*MM); float* Vim = alloc(2*MM);
  float* P    = alloc((size_t)BATCH * NHEAD * N_TOK * N_TOK);   // 134 MB
  float* AOre = alloc(2*MM); float* AOim = alloc(2*MM);
  float* FNre = alloc(2*MM); float* FNim = alloc(2*MM);
  float* pxr  = alloc(1024); float* pxi = alloc(1024);
  float* pyr  = alloc(1024); float* pyi = alloc(1024);
  float* scal = alloc(8);    // [0]=lambda_max, [1]=alpha

  // ---------------- Cayley: W = (I-A)^2 (I-A^2)^-1 for 4 matrices ----------
  for (int mtx = 0; mtx < 4; ++mtx) {
    const float* Sre = (const float*)d_in[2 + 2*mtx];
    const float* Sim = (const float*)d_in[3 + 2*mtx];

    buildA_kernel<<<4096, 256, 0, stream>>>(Sre, Sim, Are, Aim);
    launch_cgemm(Are, Aim, Are, Aim, Tre, Tim, 1024,1024,1024, 1024,1024,1024, 0, stream); // A^2
    prepare_kernel<<<4096, 256, 0, stream>>>(Are, Aim, Tre, Tim, Hre, Him, B2re, B2im);

    // Power iteration for lambda_max(H) (deterministic, fixed count)
    initx_kernel<<<4, 256, 0, stream>>>(pxr, pxi);
    for (int p = 0; p < POW_ITERS; ++p) {
      cmatvec_kernel<<<4, 256, 0, stream>>>(Hre, Him, pxr, pxi, pyr, pyi);
      normscale_kernel<<<1, 256, 0, stream>>>(pyr, pyi, pxr, pxi, scal);
    }
    alpha_kernel<<<1, 1, 0, stream>>>(scal);

    // Newton-Schulz: X <- X (2I - H X)
    initX_kernel<<<4096, 256, 0, stream>>>(Xre, Xim, scal);
    float *xcr = Xre, *xci = Xim, *xnr = X2re, *xni = X2im;
    for (int it = 0; it < NS_ITERS; ++it) {
      launch_cgemm(Hre, Him, xcr, xci, Tre, Tim, 1024,1024,1024, 1024,1024,1024, 0, stream);
      uker_kernel<<<4096, 256, 0, stream>>>(Tre, Tim, Ure, Uim);
      launch_cgemm(xcr, xci, Ure, Uim, xnr, xni, 1024,1024,1024, 1024,1024,1024, 0, stream);
      float* t;
      t = xcr; xcr = xnr; xnr = t;
      t = xci; xci = xni; xni = t;
    }
    // W = (I-A)^2 * H^-1
    launch_cgemm(B2re, B2im, xcr, xci, Wm[mtx][0], Wm[mtx][1],
                 1024,1024,1024, 1024,1024,1024, 0, stream);
  }

  // ---------------- Projections: Q/K/V = x @ W^T (complex, no conj) --------
  const int MX = BATCH * N_TOK;  // 2048
  launch_cgemm(x_re, x_im, Wm[0][0], Wm[0][1], Qre, Qim, MX,1024,1024, 1024,1024,1024, 1, stream);
  launch_cgemm(x_re, x_im, Wm[1][0], Wm[1][1], Kre, Kim, MX,1024,1024, 1024,1024,1024, 1, stream);
  launch_cgemm(x_re, x_im, Wm[2][0], Wm[2][1], Vre, Vim, MX,1024,1024, 1024,1024,1024, 1, stream);

  // ---------------- Scores p = |SCALE * conj(Q).K|^2 -----------------------
  {
    dim3 grid(N_TOK / 128, N_TOK / 16, BATCH * NHEAD);
    scores_kernel<<<grid, 256, 0, stream>>>(Qre, Qim, Kre, Kim, P);
  }

  // ---------------- Top-k + softmax + V gather -----------------------------
  topk_gather_kernel<<<BATCH * NHEAD * N_TOK, 256, 0, stream>>>(P, Vre, Vim, AOre, AOim);

  // ---------------- Output projection: out @ Wo^T --------------------------
  launch_cgemm(AOre, AOim, Wm[3][0], Wm[3][1], FNre, FNim,
               MX, 1024, 1024, 1024, 1024, 1024, 1, stream);

  // ---------------- Interleave to complex64 layout -------------------------
  interleave_kernel<<<(unsigned)(2*MM / 256), 256, 0, stream>>>(FNre, FNim, (float*)d_out);
}
